// GCN_85667417686171
// MI455X (gfx1250) — matrix operand order repaired and checked by reading, not compiled, not run on hardware
//
#include <hip/hip_runtime.h>
#include <hip/hip_bf16.h>

typedef __attribute__((ext_vector_type(16))) _Float16 v16h;
typedef __attribute__((ext_vector_type(8)))  float    v8f;
typedef __attribute__((ext_vector_type(4)))  float    v4f;

#define F_DIM 128
#define NCLS  10

// ---------------------------------------------------------------- fill
__global__ void fill_kernel(float* __restrict__ p, float v, long n) {
    long i = (long)blockIdx.x * blockDim.x + threadIdx.x;
    if (i < n) p[i] = v;
}

// ---------------------------------------------------------------- degree scatter (deg pre-filled with 1.0 for self loops)
__global__ void deg_kernel(const int* __restrict__ row, float* __restrict__ deg, int E) {
    int e = blockIdx.x * blockDim.x + threadIdx.x;
    if (e < E) atomicAdd(&deg[row[e]], 1.0f);
}

// ---------------------------------------------------------------- dinv = deg^-0.5 (in place)
__global__ void dinv_kernel(float* __restrict__ deg, int n) {
    int i = blockIdx.x * blockDim.x + threadIdx.x;
    if (i < n) deg[i] = 1.0f / sqrtf(deg[i]);
}

// ---------------------------------------------------------------- f32 -> f16 convert (weights)
__global__ void cvt_f16_kernel(const float* __restrict__ src, _Float16* __restrict__ dst, int n) {
    int i = blockIdx.x * blockDim.x + threadIdx.x;
    if (i < n) dst[i] = (_Float16)src[i];
}

// ---------------------------------------------------------------- Y[nrows,128] = X @ W^T + b  via v_wmma_f32_16x16x32_f16
// One wave handles a 16-row strip (all 8 column tiles); 8 waves / block.
// nrows must be a multiple of 16 (50000 = 3125*16).
__global__ void __launch_bounds__(256)
gemm_xwT_wmma(const float* __restrict__ X, const _Float16* __restrict__ Wh,
              const float* __restrict__ bias, float* __restrict__ Y, int nrows)
{
    const int wave    = threadIdx.x >> 5;
    const int lane    = threadIdx.x & 31;
    const int rowtile = blockIdx.x * 8 + wave;
    if (rowtile * 16 >= nrows) return;          // wave-uniform: EXEC stays all-1s

    const int m16 = lane & 15;                  // row within tile (A), col within tile (B/D)
    const int g   = lane >> 4;                  // half-wave group

    // ---- A fragments for K = 0..127 (4 chunks of 32), f32 -> f16 in regs
    v16h a[4];
    const float* xrow = X + (long)(rowtile * 16 + m16) * F_DIM;
#pragma unroll
    for (int kc = 0; kc < 4; ++kc) {
        const int k0 = kc * 32 + g * 16;
        v4f p0 = *(const v4f*)(xrow + k0);
        v4f p1 = *(const v4f*)(xrow + k0 + 4);
        v4f p2 = *(const v4f*)(xrow + k0 + 8);
        v4f p3 = *(const v4f*)(xrow + k0 + 12);
#pragma unroll
        for (int j = 0; j < 4; ++j) {
            a[kc][j]      = (_Float16)p0[j];
            a[kc][4 + j]  = (_Float16)p1[j];
            a[kc][8 + j]  = (_Float16)p2[j];
            a[kc][12 + j] = (_Float16)p3[j];
        }
    }

#pragma unroll
    for (int nt = 0; nt < 8; ++nt) {
        const int ncol = nt * 16 + m16;
        const float bv = bias[ncol];
        v8f acc;
#pragma unroll
        for (int i = 0; i < 8; ++i) acc[i] = bv;
#pragma unroll
        for (int kc = 0; kc < 4; ++kc) {
            // B fragment: 16 sequential K values of column ncol (row-major f16 W)
            v16h b = *(const v16h*)(Wh + (long)ncol * F_DIM + kc * 32 + g * 16);
            acc = __builtin_amdgcn_wmma_f32_16x16x32_f16(
                false, a[kc], false, b, (short)0, acc, false, false);
        }
        // D layout: VGPR i -> row (i + 8g), col m16
        float* yout = Y + (long)(rowtile * 16 + 8 * g) * F_DIM + ncol;
#pragma unroll
        for (int i = 0; i < 8; ++i) yout[(long)i * F_DIM] = acc[i];
    }
}

// ---------------------------------------------------------------- normalized scatter-add over edges + self loops
// 32 threads per edge, 4 floats per thread (float4 load, 4 f32 atomics)
__global__ void scatter_kernel(const float* __restrict__ Hlin, const int* __restrict__ row,
                               const int* __restrict__ col, const float* __restrict__ dinv,
                               float* __restrict__ Hout, int E, int N)
{
    long gid = (long)blockIdx.x * blockDim.x + threadIdx.x;
    long total = (long)(E + N) * 32;
    if (gid >= total) return;
    int e = (int)(gid >> 5);
    int q = ((int)gid & 31) * 4;
    int r, c;
    if (e < E) { r = row[e]; c = col[e]; } else { r = c = e - E; }
    float nrm = dinv[r] * dinv[c];
    v4f v = *(const v4f*)(Hlin + (long)r * F_DIM + q);
    float* out = Hout + (long)c * F_DIM + q;
    atomicAdd(out + 0, nrm * v[0]);
    atomicAdd(out + 1, nrm * v[1]);
    atomicAdd(out + 2, nrm * v[2]);
    atomicAdd(out + 3, nrm * v[3]);
}

// ---------------------------------------------------------------- edge head: logits + log-softmax + NLL
// one wave32 per edge; Wfc staged in LDS; out[0] = loss, out[1..] = logits [E,10]
__global__ void __launch_bounds__(256)
edge_head_kernel(const float* __restrict__ H2, const float* __restrict__ Wfc,
                 const float* __restrict__ bfc, const int* __restrict__ row,
                 const int* __restrict__ col, const int* __restrict__ label,
                 float* __restrict__ out, int E)
{
    __shared__ float sW[NCLS * 2 * F_DIM];
    __shared__ float sb[NCLS];
    __shared__ float sloss;
    for (int i = threadIdx.x; i < NCLS * 2 * F_DIM; i += 256) sW[i] = Wfc[i];
    if (threadIdx.x < NCLS) sb[threadIdx.x] = bfc[threadIdx.x];
    if (threadIdx.x == 0) sloss = 0.0f;
    __syncthreads();

    const int lane = threadIdx.x & 31;
    const int wave = threadIdx.x >> 5;
    const int nwaves = gridDim.x * 8;
    float lsum = 0.0f;

    for (int e = blockIdx.x * 8 + wave; e < E; e += nwaves) {
        const int r = row[e], c = col[e];
        float x[8];
#pragma unroll
        for (int j = 0; j < 4; ++j) {
            x[j]     = H2[(long)r * F_DIM + j * 32 + lane];
            x[4 + j] = H2[(long)c * F_DIM + j * 32 + lane];
        }
        float logit[NCLS];
#pragma unroll
        for (int cls = 0; cls < NCLS; ++cls) {
            float s = 0.0f;
#pragma unroll
            for (int j = 0; j < 8; ++j) s += x[j] * sW[cls * 256 + j * 32 + lane];
#pragma unroll
            for (int off = 16; off > 0; off >>= 1) s += __shfl_xor(s, off, 32);
            logit[cls] = s + sb[cls];
        }
        if (lane == 0) {
            float m = logit[0];
#pragma unroll
            for (int cls = 1; cls < NCLS; ++cls) m = fmaxf(m, logit[cls]);
            float se = 0.0f;
#pragma unroll
            for (int cls = 0; cls < NCLS; ++cls) se += expf(logit[cls] - m);
            float lse = m + logf(se);
            const int lab = label[e];
            float picked = 0.0f;
#pragma unroll
            for (int cls = 0; cls < NCLS; ++cls) {
                out[1 + (long)e * NCLS + cls] = logit[cls];
                if (cls == lab) picked = logit[cls];
            }
            lsum += lse - picked;
        }
    }
    if (lane == 0) atomicAdd(&sloss, lsum);
    __syncthreads();
    if (threadIdx.x == 0) atomicAdd(out, sloss / (float)E);
}

// ---------------------------------------------------------------- host
static inline size_t alignup(size_t x) { return (x + 255) & ~(size_t)255; }

extern "C" void kernel_launch(void* const* d_in, const int* in_sizes, int n_in,
                              void* d_out, int out_size, void* d_ws, size_t ws_size,
                              hipStream_t stream) {
    const float* features = (const float*)d_in[0];
    const float* W1  = (const float*)d_in[1];
    const float* b1  = (const float*)d_in[2];
    const float* W2  = (const float*)d_in[3];
    const float* b2  = (const float*)d_in[4];
    const float* Wfc = (const float*)d_in[5];
    const float* bfc = (const float*)d_in[6];
    const int* row   = (const int*)d_in[7];
    const int* col   = (const int*)d_in[8];
    const int* label = (const int*)d_in[9];

    const int N = in_sizes[0] / F_DIM;   // 50000
    const int E = in_sizes[7];           // 600000
    float* out = (float*)d_out;

    // workspace carve-up
    char* ws = (char*)d_ws;
    const size_t szH = alignup((size_t)N * F_DIM * sizeof(float));
    float* A     = (float*)ws;                         // h1lin / h2lin
    float* B     = (float*)(ws + szH);                 // h1 / h2
    float* dinv  = (float*)(ws + 2 * szH);             // deg -> dinv
    _Float16* w1h = (_Float16*)(ws + 2 * szH + alignup((size_t)N * sizeof(float)));
    _Float16* w2h = w1h + F_DIM * F_DIM;

    const int TB = 256;
    const long NH = (long)N * F_DIM;
    const int gemm_blocks = ((N + 15) / 16 + 7) / 8;
    const long scat_threads = (long)(E + N) * 32;

    // degrees & norm
    fill_kernel<<<(N + TB - 1) / TB, TB, 0, stream>>>(dinv, 1.0f, N);
    deg_kernel<<<(E + TB - 1) / TB, TB, 0, stream>>>(row, dinv, E);
    dinv_kernel<<<(N + TB - 1) / TB, TB, 0, stream>>>(dinv, N);

    // f16 weights
    cvt_f16_kernel<<<(F_DIM * F_DIM + TB - 1) / TB, TB, 0, stream>>>(W1, w1h, F_DIM * F_DIM);
    cvt_f16_kernel<<<(F_DIM * F_DIM + TB - 1) / TB, TB, 0, stream>>>(W2, w2h, F_DIM * F_DIM);

    // layer 1: A = X*W1^T + b1 ; B = scatter(A)
    gemm_xwT_wmma<<<gemm_blocks, TB, 0, stream>>>(features, w1h, b1, A, N);
    fill_kernel<<<(int)((NH + TB - 1) / TB), TB, 0, stream>>>(B, 0.0f, NH);
    scatter_kernel<<<(int)((scat_threads + TB - 1) / TB), TB, 0, stream>>>(A, row, col, dinv, B, E, N);

    // layer 2: A = B*W2^T + b2 ; B = scatter(A)   (buffer reuse: h1lin dead, then h1 dead)
    gemm_xwT_wmma<<<gemm_blocks, TB, 0, stream>>>(B, w2h, b2, A, N);
    fill_kernel<<<(int)((NH + TB - 1) / TB), TB, 0, stream>>>(B, 0.0f, NH);
    // note: scatter2 writes into B while B's old contents (h1) are no longer needed,
    // but it must start from zeros -> refill happened just above, then accumulate via atomics;
    // the fill runs after the gemm has read h1 (same stream), so h1 is intact when used.
    scatter_kernel<<<(int)((scat_threads + TB - 1) / TB), TB, 0, stream>>>(A, row, col, dinv, B, E, N);

    // edge classifier head + loss
    fill_kernel<<<1, 1, 0, stream>>>(out, 0.0f, 1);
    edge_head_kernel<<<2048, TB, 0, stream>>>(B, Wfc, bfc, row, col, label, out, E);
}
